// PetraRQSelfAttention_21577915695769
// MI455X (gfx1250) — compile-verified, matrix operand order checked by reading, not run on hardware
//
#include <hip/hip_runtime.h>

// ---------------------------------------------------------------------------
// PetraRQ (Linformer) attention for MI455X / gfx1250, bf16 WMMA pipeline.
// ---------------------------------------------------------------------------

typedef __bf16 bf16;
typedef __attribute__((ext_vector_type(16))) __bf16 v16bf;
typedef __attribute__((ext_vector_type(8)))  float  v8f;

#define WMMA_BF16(A, B, C) \
    __builtin_amdgcn_wmma_f32_16x16x32_bf16(false, (A), false, (B), (short)0, (C), false, false)

// K-element index inside a 16x32 bf16 A/B fragment (ISA 7.12.2):
//   lanes 0-15  : elements 0..7 -> K 0..7,  elements 8..15 -> K 16..23
//   lanes 16-31 : elements 0..7 -> K 8..15, elements 8..15 -> K 24..31
__device__ __forceinline__ int frag_k(int e, int half) {
    return (e & 7) + ((e >> 3) << 4) + (half << 3);
}

// ---------------------------------------------------------------------------
// Cast helpers (one-time precision conversion, bandwidth-trivial)
// ---------------------------------------------------------------------------
__global__ void cast_f32_bf16(const float* __restrict__ s, bf16* __restrict__ d, long long n) {
    long long i = (long long)blockIdx.x * blockDim.x + threadIdx.x;
    long long stride = (long long)gridDim.x * blockDim.x;
    for (; i < n; i += stride) d[i] = (bf16)s[i];
}

// s[rows][cols] f32 -> d[r*dstLd + c] bf16 (pack Wq|Wk|Wv side by side)
__global__ void cast_f32_bf16_strided(const float* __restrict__ s, bf16* __restrict__ d,
                                      int rows, int cols, int dstLd) {
    long long n = (long long)rows * cols;
    long long i = (long long)blockIdx.x * blockDim.x + threadIdx.x;
    long long stride = (long long)gridDim.x * blockDim.x;
    for (; i < n; i += stride) {
        int r = (int)(i / cols), c = (int)(i % cols);
        d[(long long)r * dstLd + c] = (bf16)s[i];
    }
}

// s[rows][cols] f32 -> d[cols][rows] bf16 (proj_k / proj_v transpose)
__global__ __launch_bounds__(256) void cast_transpose_f32_bf16(
    const float* __restrict__ s, bf16* __restrict__ d, int rows, int cols) {
    __shared__ float tile[32][33];
    const int c0 = blockIdx.x * 32, r0 = blockIdx.y * 32;
    const int tx = threadIdx.x & 31, ty = threadIdx.x >> 5;   // 32 x 8
#pragma unroll
    for (int i = 0; i < 32; i += 8)
        tile[ty + i][tx] = s[(long long)(r0 + ty + i) * cols + (c0 + tx)];
    __syncthreads();
#pragma unroll
    for (int i = 0; i < 32; i += 8)
        d[(long long)(c0 + ty + i) * rows + (r0 + tx)] = (bf16)tile[tx][ty + i];
}

// ---------------------------------------------------------------------------
// Generic bf16 WMMA GEMM: C = A @ B (+bias), 128x128 block, 8 waves x (32x64)
// ---------------------------------------------------------------------------
template <bool OUT_BF16, bool BIAS>
__global__ __launch_bounds__(256) void gemm_wmma_bf16(
    const bf16* __restrict__ A, const bf16* __restrict__ B, void* __restrict__ C,
    const float* __restrict__ bias, int M, int N, int Kd, int lda, int ldb, int ldc,
    long long strideA, long long strideB, long long strideC) {
    __shared__ bf16 As[128 * 32];
    __shared__ bf16 Bs[32 * 128];

    const int bz = blockIdx.z;
    A += (long long)bz * strideA;
    B += (long long)bz * strideB;

    const int row0 = blockIdx.y * 128;
    const int col0 = blockIdx.x * 128;
    const int t    = threadIdx.x;
    const int lane = t & 31;
    const int wave = t >> 5;
    const int wm   = wave & 3;      // 4 waves along M
    const int wn   = wave >> 2;     // 2 waves along N
    const int half = lane >> 4;
    const int l16  = lane & 15;

    v8f acc[2][4];
#pragma unroll
    for (int i = 0; i < 2; ++i)
#pragma unroll
        for (int j = 0; j < 4; ++j)
#pragma unroll
            for (int e = 0; e < 8; ++e) acc[i][j][e] = 0.0f;

    // cooperative tile-load coordinates (16 bf16 = 32B per thread per tile)
    const int ar = t >> 1, ac0 = (t & 1) * 16;   // A: 128 rows x 32 cols
    const int br = t >> 3, bc0 = (t & 7) * 16;   // B: 32 rows x 128 cols

    for (int k0 = 0; k0 < Kd; k0 += 32) {
        const uint4* gA = (const uint4*)(A + (long long)(row0 + ar) * lda + k0 + ac0);
        const uint4* gB = (const uint4*)(B + (long long)(k0 + br) * ldb + col0 + bc0);
        uint4 a0 = gA[0], a1 = gA[1];
        uint4 b0 = gB[0], b1 = gB[1];
        ((uint4*)(As + ar * 32 + ac0))[0] = a0;
        ((uint4*)(As + ar * 32 + ac0))[1] = a1;
        ((uint4*)(Bs + br * 128 + bc0))[0] = b0;
        ((uint4*)(Bs + br * 128 + bc0))[1] = b1;

        if (k0 + 32 < Kd) {  // global_prefetch_b8 the next k-tile
            __builtin_prefetch((const void*)(A + (long long)(row0 + ar) * lda + (k0 + 32) + ac0), 0, 0);
            __builtin_prefetch((const void*)(B + (long long)(k0 + 32 + br) * ldb + col0 + bc0), 0, 0);
        }
        __syncthreads();

        v16bf afrag[2];
#pragma unroll
        for (int i = 0; i < 2; ++i) {
            const bf16* p = As + (wm * 32 + i * 16 + l16) * 32;
#pragma unroll
            for (int e = 0; e < 16; ++e) afrag[i][e] = p[frag_k(e, half)];
        }
        v16bf bfrag[4];
#pragma unroll
        for (int j = 0; j < 4; ++j) {
            const int c = wn * 64 + j * 16 + l16;
#pragma unroll
            for (int e = 0; e < 16; ++e) bfrag[j][e] = Bs[frag_k(e, half) * 128 + c];
        }
#pragma unroll
        for (int i = 0; i < 2; ++i)
#pragma unroll
            for (int j = 0; j < 4; ++j)
                acc[i][j] = WMMA_BF16(afrag[i], bfrag[j], acc[i][j]);
        __syncthreads();
    }

#pragma unroll
    for (int i = 0; i < 2; ++i)
#pragma unroll
        for (int j = 0; j < 4; ++j) {
            const int c = col0 + wn * 64 + j * 16 + l16;
#pragma unroll
            for (int r = 0; r < 8; ++r) {
                const int m = row0 + wm * 32 + i * 16 + r + half * 8;
                const float v = acc[i][j][r];
                if (OUT_BF16) {
                    ((bf16*)C)[(long long)bz * strideC + (long long)m * ldc + c] = (bf16)v;
                } else {
                    float o = v;
                    if (BIAS) o += bias[c];
                    ((float*)C)[(long long)bz * strideC + (long long)m * ldc + c] = o;
                }
            }
        }
}

// ---------------------------------------------------------------------------
// Fused attention: per block one (b,h), 128 query rows, K/V resident in LDS.
// Online (flash) softmax; probs -> WMMA via per-wave LDS staging.
// ---------------------------------------------------------------------------
__global__ __launch_bounds__(256) void attn_wmma(
    const bf16* __restrict__ qkv,  // [B*4096, 3072], q in cols [0,1024)
    const bf16* __restrict__ kp,   // [B*256, 1024]
    const bf16* __restrict__ vp,   // [B*256, 1024]
    bf16* __restrict__ ao) {       // [B*4096, 1024]
    __shared__ bf16 kx[256 * 64];
    __shared__ bf16 vx[256 * 64];
    __shared__ bf16 pbuf[8][16 * 32];

    const int b = blockIdx.z, h = blockIdx.y, q0 = blockIdx.x * 128;
    const int t = threadIdx.x, lane = t & 31, wave = t >> 5;
    const int half = lane >> 4, l16 = lane & 15;

    {   // thread t loads K/V row t (64 bf16 = 128B each)
        const uint4* gk = (const uint4*)(kp + (long long)(b * 256 + t) * 1024 + h * 64);
        const uint4* gv = (const uint4*)(vp + (long long)(b * 256 + t) * 1024 + h * 64);
        uint4* sk = (uint4*)(kx + t * 64);
        uint4* sv = (uint4*)(vx + t * 64);
#pragma unroll
        for (int i = 0; i < 8; ++i) { sk[i] = gk[i]; sv[i] = gv[i]; }
    }
    __syncthreads();

    // q fragments for this wave's 16 rows (d = 0..63 -> 2 frags)
    const int qrow = q0 + wave * 16 + l16;
    const bf16* qp = qkv + (long long)(b * 4096 + qrow) * 3072 + h * 64;
    v16bf aq[2];
#pragma unroll
    for (int f = 0; f < 2; ++f)
#pragma unroll
        for (int e = 0; e < 16; ++e) aq[f][e] = qp[frag_k(e, half) + f * 32];

    v8f of[4];
#pragma unroll
    for (int j = 0; j < 4; ++j)
#pragma unroll
        for (int e = 0; e < 8; ++e) of[j][e] = 0.0f;
    float rmax[8], rsum[8];
#pragma unroll
    for (int r = 0; r < 8; ++r) { rmax[r] = -1e30f; rsum[r] = 0.0f; }

    bf16* pw = pbuf[wave];

    for (int s = 0; s < 8; ++s) {          // 256 keys in steps of 32
        const int c0 = s * 32;
        v8f d0, d1;
#pragma unroll
        for (int e = 0; e < 8; ++e) { d0[e] = 0.0f; d1[e] = 0.0f; }
#pragma unroll
        for (int f = 0; f < 2; ++f) {      // d-dim chunks 0..31, 32..63
            v16bf bk0, bk1;
#pragma unroll
            for (int e = 0; e < 16; ++e) {
                const int kk = frag_k(e, half) + f * 32;
                bk0[e] = kx[(c0 + l16) * 64 + kk];        // keys c0..c0+15
                bk1[e] = kx[(c0 + 16 + l16) * 64 + kk];   // keys c0+16..c0+31
            }
            d0 = WMMA_BF16(aq[f], bk0, d0);
            d1 = WMMA_BF16(aq[f], bk1, d1);
        }

        // online softmax update for the 8 rows this lane holds (row = r + 8*half)
#pragma unroll
        for (int r = 0; r < 8; ++r) {
            float x0 = d0[r] * 0.125f, x1 = d1[r] * 0.125f;   // DH^-0.5
            float cm = fmaxf(x0, x1);
#pragma unroll
            for (int m = 1; m < 16; m <<= 1) cm = fmaxf(cm, __shfl_xor(cm, m, 16));
            const float nm = fmaxf(rmax[r], cm);
            const float scale = __expf(rmax[r] - nm);
            const float p0 = __expf(x0 - nm), p1 = __expf(x1 - nm);
            float ps = p0 + p1;
#pragma unroll
            for (int m = 1; m < 16; m <<= 1) ps += __shfl_xor(ps, m, 16);
            rsum[r] = rsum[r] * scale + ps;
            rmax[r] = nm;
#pragma unroll
            for (int j = 0; j < 4; ++j) of[j][r] *= scale;
            pw[(r + 8 * half) * 32 + l16]      = (bf16)p0;
            pw[(r + 8 * half) * 32 + 16 + l16] = (bf16)p1;
        }
        // same-wave DS ops are in-order on CDNA5: one split-counter wait suffices
        asm volatile("s_wait_dscnt 0" ::: "memory");

        v16bf ap;   // probs as 16x32 A-fragment
#pragma unroll
        for (int e = 0; e < 16; ++e) ap[e] = pw[l16 * 32 + frag_k(e, half)];
#pragma unroll
        for (int j = 0; j < 4; ++j) {
            v16bf bv;
#pragma unroll
            for (int e = 0; e < 16; ++e)
                bv[e] = vx[(c0 + frag_k(e, half)) * 64 + j * 16 + l16];
            of[j] = WMMA_BF16(ap, bv, of[j]);
        }
    }

#pragma unroll
    for (int j = 0; j < 4; ++j)
#pragma unroll
        for (int r = 0; r < 8; ++r) {
            const int m = q0 + wave * 16 + r + 8 * half;
            const int c = h * 64 + j * 16 + l16;
            ao[(long long)(b * 4096 + m) * 1024 + c] = (bf16)(of[j][r] / rsum[r]);
        }
}

// ---------------------------------------------------------------------------
// Host-side orchestration
// ---------------------------------------------------------------------------
extern "C" void kernel_launch(void* const* d_in, const int* in_sizes, int n_in,
                              void* d_out, int out_size, void* d_ws, size_t ws_size,
                              hipStream_t stream) {
    (void)in_sizes; (void)n_in; (void)out_size; (void)ws_size;
    const float* x      = (const float*)d_in[0];
    const float* Wq     = (const float*)d_in[1];
    const float* Wk     = (const float*)d_in[2];
    const float* Wv     = (const float*)d_in[3];
    const float* proj_k = (const float*)d_in[4];
    const float* proj_v = (const float*)d_in[5];
    const float* Wo     = (const float*)d_in[6];
    const float* bo     = (const float*)d_in[7];

    const long long BN = 8LL * 4096;          // 32768 rows

    // workspace layout (bytes)
    char* ws = (char*)d_ws;
    size_t off = 0;
    bf16* xb   = (bf16*)(ws + off); off += (size_t)BN * 1024 * 2;        //  64 MB (reused as ao)
    bf16* qkv  = (bf16*)(ws + off); off += (size_t)BN * 3072 * 2;        // 192 MB
    bf16* wqkv = (bf16*)(ws + off); off += (size_t)1024 * 3072 * 2;      //   6 MB
    bf16* pkT  = (bf16*)(ws + off); off += (size_t)256 * 4096 * 2;       //   2 MB
    bf16* pvT  = (bf16*)(ws + off); off += (size_t)256 * 4096 * 2;       //   2 MB
    bf16* wob  = (bf16*)(ws + off); off += (size_t)1024 * 1024 * 2;      //   2 MB
    bf16* kpb  = (bf16*)(ws + off); off += (size_t)8 * 256 * 1024 * 2;   //   4 MB
    bf16* vpb  = (bf16*)(ws + off); off += (size_t)8 * 256 * 1024 * 2;   //   4 MB
    bf16* ao   = xb;                                                     // reuse x_bf16

    // 1) precision conversion
    cast_f32_bf16<<<8192, 256, 0, stream>>>(x, xb, BN * 1024);
    cast_f32_bf16_strided<<<2048, 256, 0, stream>>>(Wq, wqkv + 0,    1024, 1024, 3072);
    cast_f32_bf16_strided<<<2048, 256, 0, stream>>>(Wk, wqkv + 1024, 1024, 1024, 3072);
    cast_f32_bf16_strided<<<2048, 256, 0, stream>>>(Wv, wqkv + 2048, 1024, 1024, 3072);
    cast_transpose_f32_bf16<<<dim3(256 / 32, 4096 / 32), 256, 0, stream>>>(proj_k, pkT, 4096, 256);
    cast_transpose_f32_bf16<<<dim3(256 / 32, 4096 / 32), 256, 0, stream>>>(proj_v, pvT, 4096, 256);
    cast_f32_bf16<<<2048, 256, 0, stream>>>(Wo, wob, 1024LL * 1024);

    // 2) fused QKV projection: [32768,1024] @ [1024,3072] -> bf16 qkv
    gemm_wmma_bf16<true, false><<<dim3(3072 / 128, 32768 / 128, 1), 256, 0, stream>>>(
        xb, wqkv, qkv, nullptr, 32768, 3072, 1024, 1024, 3072, 3072, 0, 0, 0);

    // 3) sequence projection per batch: projT [256,4096] @ keys/values [4096,1024]
    gemm_wmma_bf16<true, false><<<dim3(1024 / 128, 256 / 128, 8), 256, 0, stream>>>(
        pkT, qkv + 1024, kpb, nullptr, 256, 1024, 4096, 4096, 3072, 1024,
        0, 4096LL * 3072, 256LL * 1024);
    gemm_wmma_bf16<true, false><<<dim3(1024 / 128, 256 / 128, 8), 256, 0, stream>>>(
        pvT, qkv + 2048, vpb, nullptr, 256, 1024, 4096, 4096, 3072, 1024,
        0, 4096LL * 3072, 256LL * 1024);

    // 4) fused attention (dots, softmax, P@V), heads recombined on store
    attn_wmma<<<dim3(4096 / 128, 16, 8), 256, 0, stream>>>(qkv, kpb, vpb, ao);

    // 5) output projection + bias -> f32
    gemm_wmma_bf16<false, true><<<dim3(1024 / 128, 32768 / 128, 1), 256, 0, stream>>>(
        ao, wob, d_out, bo, 32768, 1024, 1024, 1024, 1024, 1024, 0, 0, 0);
}
